// NonLocalTransformer_68375879353042
// MI455X (gfx1250) — matrix operand order repaired and checked by reading, not compiled, not run on hardware
//
#include <hip/hip_runtime.h>
#include <hip/hip_bf16.h>

typedef __attribute__((ext_vector_type(16))) __bf16 v16bf;
typedef __attribute__((ext_vector_type(8)))  float  v8f;

#define FLAG_BT   1   // W is [N,K] row-major (compute A @ W^T); else W is [K,N] row-major (A @ W)
#define FLAG_RELU 2

union Upk { __bf16 h[2]; unsigned int u; };

// ---------------------------------------------------------------------------
// z-offset helper: z -> (z/zdiv)*sb + (z%zdiv)*sh   (batch/head decomposition)
// ---------------------------------------------------------------------------
__device__ __forceinline__ size_t zofs(int z, int zdiv, long sb, long sh) {
  return (size_t)(z / zdiv) * (size_t)sb + (size_t)(z % zdiv) * (size_t)sh;
}

__device__ __forceinline__ unsigned int pack2(float v0, float v1) {
  Upk u; u.h[0] = (__bf16)v0; u.h[1] = (__bf16)v1;   // -> v_cvt_pk_bf16_f32
  return u.u;
}

// ---------------------------------------------------------------------------
// Generic batched GEMM: C = act(scale * (A @ op(W)) + bias) [+ R]
//   A: [M,K] fp32, row stride lda
//   W: FLAG_BT -> [N,K] (A@W^T), else [K,N] (A@W); row stride ldw
//   C: [M,N] fp32, row stride ldc; R optional residual, row stride ldr
// Block = 128 threads (4 waves). Block tile 64x64, K step 32.
// Tiles are staged in LDS as bf16, PRE-SWIZZLED into a contiguous-k fragment layout (same for A and B):
//   A frag (16x32 bf16): lane = (row%16) + 16*(k>>4),     entry = k&15
//   B frag (32x16 bf16): lane = (col%16) + 16*(k>>4),     entry = k&15
// so each fragment is ONE contiguous 32-byte LDS read per lane.
// Interior tiles take an unguarded float2 fast path (all ld/offsets are even).
// ---------------------------------------------------------------------------
__global__ __launch_bounds__(128)
void gemm_bf16_wmma(const float* __restrict__ A, long lda, long sAb, long sAh,
                    const float* __restrict__ W, long ldw, long sWb, long sWh,
                    const float* __restrict__ Bias,
                    const float* __restrict__ R, long ldr, long sRb, long sRh,
                    float* __restrict__ C, long ldc, long sCb, long sCh,
                    int M, int N, int K, int zdiv, float scale, int flags)
{
  // [strip/tile][lane][entry] : 4*32*16 bf16 = 4 KB each
  __shared__ __align__(32) __bf16 Apk[4][32][16];
  __shared__ __align__(32) __bf16 Bpk[4][32][16];

  const int tid  = threadIdx.x;
  const int lane = tid & 31;
  const int wave = tid >> 5;
  const int half = lane >> 4;   // 0 or 1 (lane group)
  const int lm   = lane & 15;

  const int z = blockIdx.z;
  const size_t offA = zofs(z, zdiv, sAb, sAh);
  const size_t offW = zofs(z, zdiv, sWb, sWh);
  const size_t offC = zofs(z, zdiv, sCb, sCh);
  const size_t offR = zofs(z, zdiv, sRb, sRh);

  const int row0 = blockIdx.y * 64;
  const int col0 = blockIdx.x * 64;
  const bool fullTile = (row0 + 64 <= M) && (col0 + 64 <= N);

  v8f zero = {};
  v8f acc[4] = {zero, zero, zero, zero};

  for (int k0 = 0; k0 < K; k0 += 32) {
    const bool fullK = (k0 + 32 <= K);
    if (fullTile && fullK) {
      // ======== fast path: unguarded, vectorized float2 loads ========
      for (int idx = tid; idx < 64 * 16; idx += 128) {
        int r  = idx >> 4;
        int kp = (idx & 15) << 1;
        float2 f = *reinterpret_cast<const float2*>(
            &A[offA + (size_t)(row0 + r) * lda + k0 + kp]);
        int h = kp >> 4;
        int e = kp & 15;
        *reinterpret_cast<unsigned int*>(&Apk[r >> 4][(r & 15) + (h << 4)][e]) =
            pack2(f.x, f.y);
      }
      if (flags & FLAG_BT) {
        for (int idx = tid; idx < 64 * 16; idx += 128) {
          int n  = idx >> 4;
          int kp = (idx & 15) << 1;
          float2 f = *reinterpret_cast<const float2*>(
              &W[offW + (size_t)(col0 + n) * ldw + k0 + kp]);
          *reinterpret_cast<unsigned int*>(
              &Bpk[n >> 4][(n & 15) + ((kp >> 4) << 4)][kp & 15]) = pack2(f.x, f.y);
        }
      } else {
        for (int idx = tid; idx < 16 * 64; idx += 128) {
          int kp = (idx >> 6) << 1;
          int n  = idx & 63;
          float v0 = W[offW + (size_t)(k0 + kp) * ldw + col0 + n];
          float v1 = W[offW + (size_t)(k0 + kp + 1) * ldw + col0 + n];
          *reinterpret_cast<unsigned int*>(
              &Bpk[n >> 4][(n & 15) + ((kp >> 4) << 4)][kp & 15]) = pack2(v0, v1);
        }
      }
    } else {
      // ======== guarded path for ragged edges (M/N/K not tile-aligned) ========
      for (int idx = tid; idx < 64 * 16; idx += 128) {
        int r  = idx >> 4;
        int kp = (idx & 15) << 1;
        int gr = row0 + r;
        int gk = k0 + kp;
        float v0 = 0.f, v1 = 0.f;
        if (gr < M) {
          if (gk < K)     v0 = A[offA + (size_t)gr * lda + gk];
          if (gk + 1 < K) v1 = A[offA + (size_t)gr * lda + gk + 1];
        }
      int h = kp >> 4;
      int e = kp & 15;
        *reinterpret_cast<unsigned int*>(&Apk[r >> 4][(r & 15) + (h << 4)][e]) =
            pack2(v0, v1);
      }
      if (flags & FLAG_BT) {
        for (int idx = tid; idx < 64 * 16; idx += 128) {
          int n  = idx >> 4;
          int kp = (idx & 15) << 1;
          int gn = col0 + n;
          int gk = k0 + kp;
          float v0 = 0.f, v1 = 0.f;
          if (gn < N) {
            if (gk < K)     v0 = W[offW + (size_t)gn * ldw + gk];
            if (gk + 1 < K) v1 = W[offW + (size_t)gn * ldw + gk + 1];
          }
          *reinterpret_cast<unsigned int*>(
              &Bpk[n >> 4][(n & 15) + ((kp >> 4) << 4)][kp & 15]) = pack2(v0, v1);
        }
      } else {
        for (int idx = tid; idx < 16 * 64; idx += 128) {
          int kp = (idx >> 6) << 1;
          int n  = idx & 63;
          int gn = col0 + n;
          int gk = k0 + kp;
          float v0 = 0.f, v1 = 0.f;
          if (gn < N) {
            if (gk < K)     v0 = W[offW + (size_t)gk * ldw + gn];
            if (gk + 1 < K) v1 = W[offW + (size_t)(gk + 1) * ldw + gn];
          }
          *reinterpret_cast<unsigned int*>(
              &Bpk[n >> 4][(n & 15) + ((kp >> 4) << 4)][kp & 15]) = pack2(v0, v1);
        }
      }
    }
    // speculative prefetch of next A k-tile (global_prefetch_b8)
    if (k0 + 32 < K) {
      int pr = row0 + (tid >> 1);
      if (pr < M) __builtin_prefetch((const void*)(A + offA + (size_t)pr * lda + k0 + 32), 0, 1);
    }
    __syncthreads();

    // ---- fragments: one contiguous 32B LDS read per operand per lane ----
    v16bf afrag = *reinterpret_cast<const v16bf*>(&Apk[wave][lane][0]);
#pragma unroll
    for (int t = 0; t < 4; ++t) {
      v16bf bfrag = *reinterpret_cast<const v16bf*>(&Bpk[t][lane][0]);
      acc[t] = __builtin_amdgcn_wmma_f32_16x16x32_bf16(
          false, afrag, false, bfrag, (short)0, acc[t], false, false);
    }
    __syncthreads();
  }

  // ---- epilogue: C layout lane=N, VGPR g -> row g + 8*(lane/16) ----
#pragma unroll
  for (int t = 0; t < 4; ++t) {
#pragma unroll
    for (int g = 0; g < 8; ++g) {
      int row = row0 + wave * 16 + half * 8 + g;
      int col = col0 + t * 16 + lm;
      if (row < M && col < N) {
        float v = acc[t][g] * scale;
        if (Bias) v += Bias[col];
        if (flags & FLAG_RELU) v = fmaxf(v, 0.f);
        if (R) v += R[offR + (size_t)row * ldr + col];
        C[offC + (size_t)row * ldc + col] = v;
      }
    }
  }
}

// ---------------------------------------------------------------------------
// Row softmax (in place). One block (256 thr) per row. Optional key mask
// (same mask vector for all rows in the launch): masked cols get -1e9.
// ---------------------------------------------------------------------------
__global__ void softmax_rows(float* __restrict__ X, int cols,
                             const unsigned char* __restrict__ mask)
{
  __shared__ float red[256];
  const int tid = threadIdx.x;
  float* row = X + (size_t)blockIdx.x * cols;

  float mx = -3.4e38f;
  for (int c = tid; c < cols; c += 256) {
    float v = row[c];
    if (mask && mask[c]) v += -1e9f;
    mx = fmaxf(mx, v);
  }
  red[tid] = mx; __syncthreads();
  for (int s = 128; s > 0; s >>= 1) {
    if (tid < s) red[tid] = fmaxf(red[tid], red[tid + s]);
    __syncthreads();
  }
  mx = red[0]; __syncthreads();

  float sum = 0.f;
  for (int c = tid; c < cols; c += 256) {
    float v = row[c];
    if (mask && mask[c]) v += -1e9f;
    float e = __expf(v - mx);
    row[c] = e;
    sum += e;
  }
  red[tid] = sum; __syncthreads();
  for (int s = 128; s > 0; s >>= 1) {
    if (tid < s) red[tid] += red[tid + s];
    __syncthreads();
  }
  float inv = 1.f / red[0];
  for (int c = tid; c < cols; c += 256) row[c] *= inv;
}

// ---------------------------------------------------------------------------
// LayerNorm over last dim D (D multiple of 32, D<=256): one wave32 per row.
// out = LN(x + res) * g + b
// ---------------------------------------------------------------------------
__global__ void ln_rows(const float* __restrict__ X, const float* __restrict__ Rres,
                        const float* __restrict__ G, const float* __restrict__ Bb,
                        float* __restrict__ O, int D)
{
  const int lane = threadIdx.x;           // blockDim = 32 (wave32)
  const size_t base = (size_t)blockIdx.x * D;
  const int per = D / 32;                 // = 8 for D=256
  float v[8];
  float s = 0.f;
  for (int i = 0; i < per; ++i) {
    int c = lane + i * 32;
    float x = X[base + c];
    if (Rres) x += Rres[base + c];
    v[i] = x; s += x;
  }
  for (int m = 16; m > 0; m >>= 1) s += __shfl_xor(s, m, 32);
  float mean = s / D;
  float q = 0.f;
  for (int i = 0; i < per; ++i) { float d = v[i] - mean; q += d * d; }
  for (int m = 16; m > 0; m >>= 1) q += __shfl_xor(q, m, 32);
  float rstd = rsqrtf(q / D + 1e-5f);
  for (int i = 0; i < per; ++i) {
    int c = lane + i * 32;
    O[base + c] = (v[i] - mean) * rstd * G[c] + Bb[c];
  }
}

// ---------------------------------------------------------------------------
// Elementwise / layout kernels
// ---------------------------------------------------------------------------
__global__ void permute_bdn_to_bnd(const float* __restrict__ S, float* __restrict__ Dst,
                                   int Bn, int Dd, int Nn)
{ // Dst[b][n][d] = S[b][d][n]
  size_t i = (size_t)blockIdx.x * blockDim.x + threadIdx.x;
  size_t total = (size_t)Bn * Dd * Nn;
  if (i >= total) return;
  int d = (int)(i % Dd); size_t t = i / Dd;
  int n = (int)(t % Nn); int b = (int)(t / Nn);
  Dst[i] = S[((size_t)b * Dd + d) * Nn + n];
}

__global__ void permute_bdn_to_nbd(const float* __restrict__ S, float* __restrict__ Dst,
                                   int Bn, int Dd, int Nn)
{ // Dst[n][b][d] = S[b][d][n]
  size_t i = (size_t)blockIdx.x * blockDim.x + threadIdx.x;
  size_t total = (size_t)Bn * Dd * Nn;
  if (i >= total) return;
  int d = (int)(i % Dd); size_t t = i / Dd;
  int b = (int)(t % Bn); int n = (int)(t / Bn);
  Dst[i] = S[((size_t)b * Dd + d) * Nn + n];
}

__global__ void add_vec(const float* __restrict__ A, const float* __restrict__ B2,
                        float* __restrict__ C2, size_t n)
{
  size_t i = (size_t)blockIdx.x * blockDim.x + threadIdx.x;
  if (i < n) C2[i] = A[i] + B2[i];
}

__global__ void add_qpos(const float* __restrict__ T, const float* __restrict__ QE,
                         float* __restrict__ O, int L, int Bn, int Dd)
{ // O[l][b][d] = T[l][b][d] + QE[l][d]
  size_t i = (size_t)blockIdx.x * blockDim.x + threadIdx.x;
  size_t total = (size_t)L * Bn * Dd;
  if (i >= total) return;
  int d = (int)(i % Dd);
  int l = (int)(i / ((size_t)Bn * Dd));
  O[i] = T[i] + QE[(size_t)l * Dd + d];
}

__global__ void copy_hs(const float* __restrict__ T, float* __restrict__ O,
                        int L, int Bn, int Dd)
{ // O[b][l][d] = T[l][b][d]
  size_t i = (size_t)blockIdx.x * blockDim.x + threadIdx.x;
  size_t total = (size_t)L * Bn * Dd;
  if (i >= total) return;
  int d = (int)(i % Dd); size_t t = i / Dd;
  int l = (int)(t % L);  int b = (int)(t / L);
  O[i] = T[(((size_t)l * Bn) + b) * Dd + d];
}

__global__ void copy_mem(const float* __restrict__ Mm, float* __restrict__ O,
                         int Bn, int Dd, int Nn)
{ // O[b][d][n] = Mm[n][b][d]
  size_t i = (size_t)blockIdx.x * blockDim.x + threadIdx.x;
  size_t total = (size_t)Bn * Dd * Nn;
  if (i >= total) return;
  int n = (int)(i % Nn); size_t t = i / Nn;
  int d = (int)(t % Dd); int b = (int)(t / Dd);
  O[i] = Mm[(((size_t)n * Bn) + b) * Dd + d];
}

// ---------------------------------------------------------------------------
// Host orchestration
// ---------------------------------------------------------------------------
static inline void gemm(hipStream_t st,
    const float* A, long lda, const float* W, long ldw, const float* bias,
    float* C, long ldc, int M, int N, int K, float scale, int flags,
    const float* R = nullptr, long ldr = 0,
    int Z = 1, int zdiv = 1,
    long sAb = 0, long sAh = 0, long sWb = 0, long sWh = 0,
    long sCb = 0, long sCh = 0, long sRb = 0, long sRh = 0)
{
  dim3 g((N + 63) / 64, (M + 63) / 64, Z), b(128);
  gemm_bf16_wmma<<<g, b, 0, st>>>(A, lda, sAb, sAh, W, ldw, sWb, sWh, bias,
                                  R, ldr, sRb, sRh, C, ldc, sCb, sCh,
                                  M, N, K, zdiv, scale, flags);
}

static inline dim3 g1(size_t n) { return dim3((unsigned)((n + 255) / 256)); }

extern "C" void kernel_launch(void* const* d_in, const int* in_sizes, int n_in,
                              void* d_out, int out_size, void* d_ws, size_t ws_size,
                              hipStream_t stream)
{
  (void)in_sizes; (void)n_in; (void)out_size; (void)ws_size;
  constexpr int Bt = 32, Dm = 256, NHd = 8, HD = 32, NLy = 6, DFFd = 2048,
                CIc = 128, Nn = 1024, NQq = 100;
  const float inv_sqrt_ci = 0.08838834764831845f;   // 1/sqrt(128)
  const float inv_sqrt_hd = 0.17677669529663687f;   // 1/sqrt(32)

  const float* src   = (const float*)d_in[0];
  const unsigned char* mask = (const unsigned char*)d_in[1];
  const float* qe    = (const float*)d_in[2];
  const float* pos   = (const float*)d_in[3];
  const float* nl_g_w = (const float*)d_in[4];
  const float* nl_g_b = (const float*)d_in[5];
  const float* nl_th_w = (const float*)d_in[6];
  const float* nl_th_b = (const float*)d_in[7];
  const float* nl_ph_w = (const float*)d_in[8];
  const float* nl_ph_b = (const float*)d_in[9];
  const float* nl_o_w = (const float*)d_in[10];
  const float* nl_o_b = (const float*)d_in[11];
  const float* sa_w  = (const float*)d_in[12];
  const float* sa_b  = (const float*)d_in[13];
  const float* sa_ow = (const float*)d_in[14];
  const float* sa_ob = (const float*)d_in[15];
  const float* ca_w  = (const float*)d_in[16];
  const float* ca_b  = (const float*)d_in[17];
  const float* ca_ow = (const float*)d_in[18];
  const float* ca_ob = (const float*)d_in[19];
  const float* l1_w  = (const float*)d_in[20];
  const float* l1_b  = (const float*)d_in[21];
  const float* l2_w  = (const float*)d_in[22];
  const float* l2_b  = (const float*)d_in[23];
  const float* n1_g  = (const float*)d_in[24];
  const float* n1_b  = (const float*)d_in[25];
  const float* n2_g  = (const float*)d_in[26];
  const float* n2_b  = (const float*)d_in[27];
  const float* n3_g  = (const float*)d_in[28];
  const float* n3_b  = (const float*)d_in[29];
  const float* fn_g  = (const float*)d_in[30];
  const float* fn_b  = (const float*)d_in[31];

  // ---- workspace carve (floats) ----
  size_t p = 0;
  float* wsf = (float*)d_ws;
  auto carve = [&](size_t n) { float* r = wsf + p; p += n; return r; };
  float* xf   = carve((size_t)Bt * Nn * Dm);
  float* posf = carve((size_t)Nn * Bt * Dm);
  float* mem  = carve((size_t)Nn * Bt * Dm);
  float* mk   = carve((size_t)Nn * Bt * Dm);
  float* th   = carve((size_t)Nn * CIc);
  float* ph   = carve((size_t)Nn * CIc);
  float* gg   = carve((size_t)Nn * CIc);
  float* pw   = carve((size_t)Nn * Nn);
  float* yb   = carve((size_t)Nn * CIc);
  float* tout = carve((size_t)NQq * Bt * Dm);
  float* q1   = carve((size_t)NQq * Bt * Dm);
  float* pq   = carve((size_t)NQq * Bt * Dm);
  float* pk   = carve((size_t)NQq * Bt * Dm);
  float* pv   = carve((size_t)NQq * Bt * Dm);
  float* att  = carve((size_t)NQq * Bt * Dm);
  float* kmem = carve((size_t)Nn * Bt * Dm);
  float* vmem = carve((size_t)Nn * Bt * Dm);
  float* scs  = carve((size_t)Bt * NHd * NQq * NQq);
  float* scx  = carve((size_t)NHd * NQq * Nn);
  float* ffh  = carve((size_t)NQq * Bt * DFFd);
  float* fft  = carve((size_t)NQq * Bt * Dm);

  const size_t tokTotal = (size_t)Bt * Dm * Nn;
  permute_bdn_to_bnd<<<g1(tokTotal), 256, 0, stream>>>(src, xf, Bt, Dm, Nn);
  permute_bdn_to_nbd<<<g1(tokTotal), 256, 0, stream>>>(pos, posf, Bt, Dm, Nn);

  // ================= NonLocal2d encoder (per-batch) =================
  for (int b = 0; b < Bt; ++b) {
    const float* xb = xf + (size_t)b * Nn * Dm;
    gemm(stream, xb, Dm, nl_th_w, Dm, nl_th_b, th, CIc, Nn, CIc, Dm, 1.f, FLAG_BT);
    gemm(stream, xb, Dm, nl_ph_w, Dm, nl_ph_b, ph, CIc, Nn, CIc, Dm, 1.f, FLAG_BT);
    gemm(stream, xb, Dm, nl_g_w,  Dm, nl_g_b,  gg, CIc, Nn, CIc, Dm, 1.f, FLAG_BT);
    // pw = theta @ phi^T * CI^-0.5, softmax, y = pw @ g
    gemm(stream, th, CIc, ph, CIc, nullptr, pw, Nn, Nn, Nn, CIc, inv_sqrt_ci, FLAG_BT);
    softmax_rows<<<Nn, 256, 0, stream>>>(pw, Nn, nullptr);
    gemm(stream, pw, Nn, gg, CIc, nullptr, yb, CIc, Nn, CIc, Nn, 1.f, 0);
    // enc = xf + y @ out_w^T + out_b, written as memory[n][b][:]
    gemm(stream, yb, CIc, nl_o_w, CIc, nl_o_b, mem + (size_t)b * Dm, (long)Bt * Dm,
         Nn, Dm, CIc, 1.f, FLAG_BT, xb, Dm);
  }
  add_vec<<<g1((size_t)Nn * Bt * Dm), 256, 0, stream>>>(mem, posf, mk, (size_t)Nn * Bt * Dm);
  hipMemsetAsync(tout, 0, (size_t)NQq * Bt * Dm * sizeof(float), stream);

  // ================= Transformer decoder =================
  const int LB = NQq * Bt;
  for (int il = 0; il < NLy; ++il) {
    // ---- self attention: mha(q, q, out) ----
    const float* Wsa = sa_w + (size_t)il * 3 * Dm * Dm;
    const float* bsa = sa_b + (size_t)il * 3 * Dm;
    add_qpos<<<g1((size_t)LB * Dm), 256, 0, stream>>>(tout, qe, q1, NQq, Bt, Dm);
    gemm(stream, q1,   Dm, Wsa,               Dm, bsa,          pq, Dm, LB, Dm, Dm, 1.f, FLAG_BT);
    gemm(stream, q1,   Dm, Wsa + Dm * Dm,     Dm, bsa + Dm,     pk, Dm, LB, Dm, Dm, 1.f, FLAG_BT);
    gemm(stream, tout, Dm, Wsa + 2 * Dm * Dm, Dm, bsa + 2 * Dm, pv, Dm, LB, Dm, Dm, 1.f, FLAG_BT);
    // scores[b,h,l,s] = qp . kp, batched over z = b*NH+h
    gemm(stream, pq, (long)Bt * Dm, pk, (long)Bt * Dm, nullptr, scs, NQq,
         NQq, NQq, HD, inv_sqrt_hd, FLAG_BT, nullptr, 0,
         Bt * NHd, NHd, Dm, HD, Dm, HD, (long)NHd * NQq * NQq, (long)NQq * NQq);
    softmax_rows<<<Bt * NHd * NQq, 256, 0, stream>>>(scs, NQq, nullptr);
    // att = scores @ vp  (B is [K,N] = vp slice)
    gemm(stream, scs, NQq, pv, (long)Bt * Dm, nullptr, att, (long)Bt * Dm,
         NQq, HD, NQq, 1.f, 0, nullptr, 0,
         Bt * NHd, NHd, (long)NHd * NQq * NQq, (long)NQq * NQq, Dm, HD, Dm, HD);
    gemm(stream, att, Dm, sa_ow + (size_t)il * Dm * Dm, Dm, sa_ob + (size_t)il * Dm,
         q1, Dm, LB, Dm, Dm, 1.f, FLAG_BT);
    ln_rows<<<LB, 32, 0, stream>>>(q1, tout, n1_g + (size_t)il * Dm, n1_b + (size_t)il * Dm, tout, Dm);

    // ---- cross attention: mha(out+qpos, memory+posf, memory) ----
    const float* Wca = ca_w + (size_t)il * 3 * Dm * Dm;
    const float* bca = ca_b + (size_t)il * 3 * Dm;
    add_qpos<<<g1((size_t)LB * Dm), 256, 0, stream>>>(tout, qe, q1, NQq, Bt, Dm);
    gemm(stream, q1,  Dm, Wca,               Dm, bca,          pq,   Dm, LB,      Dm, Dm, 1.f, FLAG_BT);
    gemm(stream, mk,  Dm, Wca + Dm * Dm,     Dm, bca + Dm,     kmem, Dm, Nn * Bt, Dm, Dm, 1.f, FLAG_BT);
    gemm(stream, mem, Dm, Wca + 2 * Dm * Dm, Dm, bca + 2 * Dm, vmem, Dm, Nn * Bt, Dm, Dm, 1.f, FLAG_BT);
    for (int b = 0; b < Bt; ++b) {
      gemm(stream, pq + (size_t)b * Dm, (long)Bt * Dm, kmem + (size_t)b * Dm, (long)Bt * Dm,
           nullptr, scx, Nn, NQq, Nn, HD, inv_sqrt_hd, FLAG_BT, nullptr, 0,
           NHd, 1, HD, 0, HD, 0, (long)NQq * Nn, 0);
      softmax_rows<<<NHd * NQq, 256, 0, stream>>>(scx, Nn, mask + (size_t)b * Nn);
      gemm(stream, scx, Nn, vmem + (size_t)b * Dm, (long)Bt * Dm, nullptr,
           att + (size_t)b * Dm, (long)Bt * Dm, NQq, HD, Nn, 1.f, 0, nullptr, 0,
           NHd, 1, (long)NQq * Nn, 0, HD, 0, HD, 0);
    }
    gemm(stream, att, Dm, ca_ow + (size_t)il * Dm * Dm, Dm, ca_ob + (size_t)il * Dm,
         q1, Dm, LB, Dm, Dm, 1.f, FLAG_BT);
    ln_rows<<<LB, 32, 0, stream>>>(q1, tout, n2_g + (size_t)il * Dm, n2_b + (size_t)il * Dm, tout, Dm);

    // ---- FFN ----
    gemm(stream, tout, Dm, l1_w + (size_t)il * DFFd * Dm, Dm, l1_b + (size_t)il * DFFd,
         ffh, DFFd, LB, DFFd, Dm, 1.f, FLAG_BT | FLAG_RELU);
    gemm(stream, ffh, DFFd, l2_w + (size_t)il * Dm * DFFd, DFFd, l2_b + (size_t)il * Dm,
         fft, Dm, LB, Dm, DFFd, 1.f, FLAG_BT);
    ln_rows<<<LB, 32, 0, stream>>>(fft, tout, n3_g + (size_t)il * Dm, n3_b + (size_t)il * Dm, tout, Dm);
  }
  // final LN + outputs
  ln_rows<<<LB, 32, 0, stream>>>(tout, nullptr, fn_g, fn_b, tout, Dm);
  float* out_f = (float*)d_out;
  copy_hs<<<g1((size_t)LB * Dm), 256, 0, stream>>>(tout, out_f, NQq, Bt, Dm);
  copy_mem<<<g1(tokTotal), 256, 0, stream>>>(mem, out_f + (size_t)LB * Dm, Bt, Dm, Nn);
}
